// EquivariantGraphNetwork_2894807957839
// MI455X (gfx1250) — matrix operand order repaired and checked by reading, not compiled, not run on hardware
//
#include <hip/hip_runtime.h>

#define Bb 4
#define Nn 512
#define Hh 128
#define Ll 4
#define RB 16

typedef __attribute__((ext_vector_type(16))) _Float16 v16h;
typedef __attribute__((ext_vector_type(8)))  _Float16 v8h;
typedef __attribute__((ext_vector_type(8)))  float    v8f;

__device__ __forceinline__ v16h cat8(v8h lo, v8h hi) {
  return __builtin_shufflevector(lo, hi, 0,1,2,3,4,5,6,7,8,9,10,11,12,13,14,15);
}

// A fragment (16x32 f16, M x K). LDS tile row-major [row][k], row stride 128 halves.
// Per ISA: lanes 0-15 hold K = k0+{0..7, 16..23}; lanes 16-31 hold K = k0+{8..15, 24..31}.
__device__ __forceinline__ v16h load_A_lds(const _Float16* base, int row, int k0, int lh) {
  const _Float16* p = base + row * Hh + k0 + 8 * lh;
  return cat8(*(const v8h*)p, *(const v8h*)(p + 16));
}

// B fragment (32x16 f16, K x N) from transposed weights wT[n][k] in LDS (row stride 128 halves).
// Per ISA: lanes 0-15 (N=lane) hold K = k0+{0..7, 16..23}; lanes 16-31 hold K = k0+{8..15, 24..31}.
__device__ __forceinline__ v16h load_B_lds(const _Float16* wT, int n, int k0, int lh) {
  const _Float16* p = wT + n * Hh + k0 + 8 * lh;
  return cat8(*(const v8h*)p, *(const v8h*)(p + 16));
}

__device__ __forceinline__ float silu_f(float v) {
  return v * __builtin_amdgcn_rcpf(1.0f + __expf(-v));
}

// ---------------- time embedding + MLP -> t_emb (B,H) ----------------
__global__ void time_init_kernel(const float* __restrict__ t,
                                 const float* __restrict__ tw1, const float* __restrict__ tb1,
                                 const float* __restrict__ tw2, const float* __restrict__ tb2,
                                 float* __restrict__ temb) {
  __shared__ float te[Bb][64];
  __shared__ float hid[Bb][Hh];
  const int tid = threadIdx.x;            // 256 threads
  {
    int b = tid >> 6, d = tid & 63;
    float fr = __expf((float)(d & 31) * -0.29710775f);   // exp(idx * -ln(1e4)/31)
    float ang = t[b] * fr;
    te[b][d] = (d < 32) ? __sinf(ang) : __cosf(ang);
  }
  __syncthreads();
  for (int it = 0; it < 2; ++it) {
    int q = tid + it * 256;               // 0..511 -> (b,k)
    int b = q >> 7, k = q & 127;
    float v = tb1[k];
    for (int m = 0; m < 64; ++m) v = fmaf(te[b][m], tw1[m * Hh + k], v);
    hid[b][k] = silu_f(v);
  }
  __syncthreads();
  for (int it = 0; it < 2; ++it) {
    int q = tid + it * 256;
    int b = q >> 7, k = q & 127;
    float v = tb2[k];
    for (int m = 0; m < Hh; ++m) v = fmaf(hid[b][m], tw2[m * Hh + k], v);
    temb[b * Hh + k] = v;
  }
}

// ---------------- h0 broadcast + zero output ----------------
__global__ void bcast_h_kernel(const float* __restrict__ temb,
                               float* __restrict__ h, float* __restrict__ out) {
  int idx = blockIdx.x * 256 + threadIdx.x;     // B*N*H = 262144
  int b = idx >> 16;                            // N*H = 65536
  int k = idx & 127;
  h[idx] = temb[b * Hh + k];
  if (idx < Bb * Nn * 3) out[idx] = 0.0f;
}

// ---------------- convert msg_w2 / cw1 to f16, transposed [l][n][k] ----------------
__global__ void conv_w_kernel(const float* __restrict__ w2, const float* __restrict__ cw1,
                              _Float16* __restrict__ W2T, _Float16* __restrict__ C1T) {
  int idx = blockIdx.x * 256 + threadIdx.x;     // L*H*H = 65536
  int l = idx >> 14;
  int nk = idx & 16383;
  int n = nk >> 7, k = nk & 127;
  W2T[idx] = (_Float16)w2[l * Hh * Hh + k * Hh + n];
  C1T[idx] = (_Float16)cw1[l * Hh * Hh + k * Hh + n];
}

// ---------------- per-layer: Ha = h@W1[a]+b1, Hb = h@W1[b] ----------------
__global__ void prep_hab_kernel(const float* __restrict__ h,
                                const float* __restrict__ w1,   // (272,128) slice of layer l
                                const float* __restrict__ b1,
                                float* __restrict__ Ha, float* __restrict__ Hb) {
  __shared__ float hrow[Hh];
  const int row = blockIdx.x;                 // B*N rows
  const int k = threadIdx.x;                  // 128 threads
  hrow[k] = h[(long)row * Hh + k];
  __syncthreads();
  float a = b1[k], bb = 0.0f;
  #pragma unroll 4
  for (int m = 0; m < Hh; ++m) {
    float hv = hrow[m];
    a  = fmaf(hv, w1[m * Hh + k], a);
    bb = fmaf(hv, w1[(Hh + m) * Hh + k], bb);
  }
  Ha[(long)row * Hh + k] = a;
  Hb[(long)row * Hh + k] = bb;
}

// LDS layout (dynamic shared memory), bytes:
//   abuf  : 8 waves * 16*128 f16  = 32768
//   sW2T  : 128*128 f16           = 32768
//   sC1T  : 128*128 f16           = 32768
//   sW1r  : 16*128 f32            =  8192
//   sHai/sB2/sCb1/sCw2/hsum : 5*128 f32 = 2560
//   emask : 8*16 f32              =   512
//   edir  : 8*16*3 f32            =  1536
//   csum  : 4 f32                 =    16
static constexpr unsigned MSG_SMEM_BYTES =
    32768u * 3 + 8192u + 2560u + 512u + 1536u + 16u;   // 111120

// ---------------- main per-layer message-passing kernel ----------------
__launch_bounds__(256, 2)
__global__ void msg_kernel(const float* __restrict__ x,
                           const float* __restrict__ h_cur,
                           const float* __restrict__ Ha,
                           const float* __restrict__ Hbuf,
                           const _Float16* __restrict__ W2Tg,  // [n][k] f16 (global)
                           const _Float16* __restrict__ C1Tg,  // [n][k] f16 (global)
                           const float* __restrict__ W1r,      // (16,128)
                           const float* __restrict__ b2,
                           const float* __restrict__ cb1,
                           const float* __restrict__ cw2,
                           const float* __restrict__ cb2,
                           float* __restrict__ h_next,
                           float* __restrict__ coord_out) {
  extern __shared__ __align__(16) char smem_raw[];
  _Float16* abuf = (_Float16*)smem_raw;          // 8 * 2048 halves
  _Float16* sW2T = abuf + 8 * 2048;              // 16384 halves
  _Float16* sC1T = sW2T + 16384;                 // 16384 halves
  float* sW1r = (float*)(sC1T + 16384);          // 2048 f
  float* sHai = sW1r + 2048;                     // 128
  float* sB2  = sHai + 128;                      // 128
  float* sCb1 = sB2 + 128;                       // 128
  float* sCw2 = sCb1 + 128;                      // 128
  float* hsum = sCw2 + 128;                      // 128
  float* emask = hsum + 128;                     // 8*16
  float* edir  = emask + 128;                    // 8*16*3
  float* csum  = edir + 384;                     // 3

  const int tid = threadIdx.x;
  const int w = tid >> 5;        // wave id 0..7
  const int lane = tid & 31;
  const int lr = lane & 15;
  const int lh = lane >> 4;

  const int bg = blockIdx.x;     // b*N + i
  const int b = bg >> 9;
  const long bi = (long)bg * Hh;

  // ---- setup: stage weights + per-node vectors into LDS ----
  if (tid < Hh) {
    sHai[tid] = Ha[bi + tid];
    hsum[tid] = h_cur[bi + tid];      // h_next = h + sum_j messages
    sB2[tid]  = b2[tid];
    sCb1[tid] = cb1[tid];
    sCw2[tid] = cw2[tid];
  }
  for (int idx = tid; idx < 2048; idx += 256) {        // 32KB each, 16B chunks
    ((float4*)sW2T)[idx] = ((const float4*)W2Tg)[idx];
    ((float4*)sC1T)[idx] = ((const float4*)C1Tg)[idx];
  }
  for (int idx = tid; idx < RB * Hh; idx += 256) sW1r[idx] = W1r[idx];
  if (tid < 3) csum[tid] = 0.0f;
  const float cb2v = cb2[0];
  const float xi0 = x[bg * 3 + 0], xi1 = x[bg * 3 + 1], xi2 = x[bg * 3 + 2];
  __syncthreads();

  for (int pass = 0; pass < 4; ++pass) {
    // compiler barrier: keep all loads inside the loop (no LICM -> no VGPR blowup/spills)
    asm volatile("" ::: "memory");

    const int jrow = pass * 128 + w * 16 + lr;    // this lane's j (both halves: same j)
    const long jbase = (long)b * Nn + jrow;

    // ---- edge geometry (redundant across halves; lh==0 stores) ----
    float dx = x[jbase * 3 + 0] - xi0;
    float dy = x[jbase * 3 + 1] - xi1;
    float dz = x[jbase * 3 + 2] - xi2;
    float dist = sqrtf(dx * dx + dy * dy + dz * dz);
    float mask = (dist < 10.0f && dist > 0.0f) ? 1.0f : 0.0f;
    float inv = __builtin_amdgcn_rcpf(dist + 1e-8f);
    if (lh == 0) {
      emask[w * 16 + lr] = mask;
      edir[(w * 16 + lr) * 3 + 0] = dx * inv;
      edir[(w * 16 + lr) * 3 + 1] = dy * inv;
      edir[(w * 16 + lr) * 3 + 2] = dz * inv;
    }
    float rbf[RB];
    #pragma unroll
    for (int r = 0; r < RB; ++r) {
      float dd = dist - (float)r * 0.66666667f;   // centers: linspace(0,10,16)
      rbf[r] = __expf(-dd * dd * 1.28f);          // 1/(2*width^2), width = 10/16
    }

    // ---- a1 = silu(Ha_i + Hb_j + rbf@W1r) -> f16 LDS tile; lane covers k in [64*lh,64*lh+64) ----
    const float* hbj = Hbuf + jbase * Hh;
    _Float16* arow = abuf + w * 2048 + lr * Hh;
    for (int kc = 0; kc < 64; kc += 8) {
      const int k = 64 * lh + kc;
      const float4 hb0 = *(const float4*)(hbj + k);
      const float4 hb1 = *(const float4*)(hbj + k + 4);
      float v[8] = { sHai[k+0]+hb0.x, sHai[k+1]+hb0.y, sHai[k+2]+hb0.z, sHai[k+3]+hb0.w,
                     sHai[k+4]+hb1.x, sHai[k+5]+hb1.y, sHai[k+6]+hb1.z, sHai[k+7]+hb1.w };
      #pragma unroll
      for (int r = 0; r < RB; ++r) {
        const float rv = rbf[r];
        const float4 w0 = *(const float4*)&sW1r[r * Hh + k];
        const float4 w1 = *(const float4*)&sW1r[r * Hh + k + 4];
        v[0] = fmaf(rv, w0.x, v[0]); v[1] = fmaf(rv, w0.y, v[1]);
        v[2] = fmaf(rv, w0.z, v[2]); v[3] = fmaf(rv, w0.w, v[3]);
        v[4] = fmaf(rv, w1.x, v[4]); v[5] = fmaf(rv, w1.y, v[5]);
        v[6] = fmaf(rv, w1.z, v[6]); v[7] = fmaf(rv, w1.w, v[7]);
      }
      v8h hv;
      #pragma unroll
      for (int u = 0; u < 8; ++u) hv[u] = (_Float16)silu_f(v[u]);
      *(v8h*)(arow + k) = hv;
    }

    // ---- GEMM1: messages(16x128) = a1(16x128) @ W2 ----
    v8f acc[8] = {};
    #pragma unroll
    for (int kt = 0; kt < 4; ++kt) {
      const v16h Af = load_A_lds(abuf + w * 2048, lr, kt * 32, lh);
      #pragma unroll
      for (int nt = 0; nt < 8; ++nt) {
        const v16h Bf = load_B_lds(sW2T, nt * 16 + lr, kt * 32, lh);
        acc[nt] = __builtin_amdgcn_wmma_f32_16x16x32_f16(
            false, Af, false, Bf, (short)0, acc[nt], false, false);
      }
    }

    // C/D layout: acc[nt] reg r -> (row = r + 8*lh, col = nt*16 + lr)
    float mrow[8];
    #pragma unroll
    for (int r = 0; r < 8; ++r) mrow[r] = emask[w * 16 + r + 8 * lh];

    #pragma unroll
    for (int nt = 0; nt < 8; ++nt) {
      const float bn = sB2[nt * 16 + lr];
      float colsum = 0.0f;
      #pragma unroll
      for (int r = 0; r < 8; ++r) {
        float m = (acc[nt][r] + bn) * mrow[r];
        colsum += m;
        abuf[w * 2048 + (r + 8 * lh) * Hh + nt * 16 + lr] = (_Float16)m;  // re-stage masked msg
      }
      atomicAdd(&hsum[nt * 16 + lr], colsum);
    }

    // ---- GEMM2: c1pre(16x128) = msg(16x128) @ cw1 ----
    v8f acc2[8] = {};
    #pragma unroll
    for (int kt = 0; kt < 4; ++kt) {
      const v16h Af = load_A_lds(abuf + w * 2048, lr, kt * 32, lh);
      #pragma unroll
      for (int nt = 0; nt < 8; ++nt) {
        const v16h Bf = load_B_lds(sC1T, nt * 16 + lr, kt * 32, lh);
        acc2[nt] = __builtin_amdgcn_wmma_f32_16x16x32_f16(
            false, Af, false, Bf, (short)0, acc2[nt], false, false);
      }
    }

    // ---- per-edge weight: silu(c1pre + cb1) . cw2 + cb2 ----
    float wpart[8] = {};
    #pragma unroll
    for (int nt = 0; nt < 8; ++nt) {
      const float c1b = sCb1[nt * 16 + lr];
      const float cwn = sCw2[nt * 16 + lr];
      #pragma unroll
      for (int r = 0; r < 8; ++r)
        wpart[r] = fmaf(silu_f(acc2[nt][r] + c1b), cwn, wpart[r]);
    }
    #pragma unroll
    for (int m = 8; m >= 1; m >>= 1) {
      #pragma unroll
      for (int r = 0; r < 8; ++r) wpart[r] += __shfl_xor(wpart[r], m, 16);
    }
    if (lr == 0) {
      #pragma unroll
      for (int r = 0; r < 8; ++r) {
        const int row = r + 8 * lh;
        const float wv = (wpart[r] + cb2v) * emask[w * 16 + row];
        atomicAdd(&csum[0], wv * edir[(w * 16 + row) * 3 + 0]);
        atomicAdd(&csum[1], wv * edir[(w * 16 + row) * 3 + 1]);
        atomicAdd(&csum[2], wv * edir[(w * 16 + row) * 3 + 2]);
      }
    }
  }

  __syncthreads();
  if (tid < Hh) h_next[bi + tid] = hsum[tid];
  if (tid < 3) coord_out[bg * 3 + tid] += csum[tid];
}

extern "C" void kernel_launch(void* const* d_in, const int* in_sizes, int n_in,
                              void* d_out, int out_size, void* d_ws, size_t ws_size,
                              hipStream_t stream) {
  (void)in_sizes; (void)n_in; (void)out_size; (void)ws_size;
  const float* x   = (const float*)d_in[0];
  const float* t   = (const float*)d_in[1];
  const float* tw1 = (const float*)d_in[2];
  const float* tb1 = (const float*)d_in[3];
  const float* tw2 = (const float*)d_in[4];
  const float* tb2 = (const float*)d_in[5];
  const float* mw1 = (const float*)d_in[6];   // (L,272,128)
  const float* mb1 = (const float*)d_in[7];   // (L,128)
  const float* mw2 = (const float*)d_in[8];   // (L,128,128)
  const float* mb2 = (const float*)d_in[9];   // (L,128)
  const float* cw1 = (const float*)d_in[10];  // (L,128,128)
  const float* cb1 = (const float*)d_in[11];  // (L,128)
  const float* cw2 = (const float*)d_in[12];  // (L,128,1)
  const float* cb2 = (const float*)d_in[13];  // (L,1)
  float* out = (float*)d_out;

  // workspace layout (floats): hA | hB | Ha | Hb | temb | [f16] W2T | C1T
  float* f   = (float*)d_ws;
  float* hA  = f;                       // 262144
  float* hB  = f + 262144;              // 262144
  float* Ha  = f + 524288;              // 262144
  float* Hb  = f + 786432;              // 262144
  float* temb = f + 1048576;            // 512
  _Float16* W2T = (_Float16*)(f + 1048576 + 512);   // L*128*128 halves, 16B-aligned
  _Float16* C1T = W2T + Ll * Hh * Hh;

  time_init_kernel<<<1, 256, 0, stream>>>(t, tw1, tb1, tw2, tb2, temb);
  bcast_h_kernel<<<1024, 256, 0, stream>>>(temb, hA, out);
  conv_w_kernel<<<256, 256, 0, stream>>>(mw2, cw1, W2T, C1T);

  float* hc = hA;
  float* hn = hB;
  for (int l = 0; l < Ll; ++l) {
    prep_hab_kernel<<<Bb * Nn, 128, 0, stream>>>(hc, mw1 + l * 272 * Hh, mb1 + l * Hh, Ha, Hb);
    msg_kernel<<<Bb * Nn, 256, MSG_SMEM_BYTES, stream>>>(
        x, hc, Ha, Hb,
        W2T + l * Hh * Hh, C1T + l * Hh * Hh,
        mw1 + l * 272 * Hh + 256 * Hh,   // rbf rows 256..271 of msg_w1[l]
        mb2 + l * Hh, cb1 + l * Hh, cw2 + l * Hh, cb2 + l,
        hn, out);
    float* tmp = hc; hc = hn; hn = tmp;
  }
}
